// FlashAttention_32263794327900
// MI455X (gfx1250) — matrix operand order repaired and checked by reading, not compiled, not run on hardware
//
#include <hip/hip_runtime.h>
#include <hip/hip_bf16.h>

typedef _Float16 v16h __attribute__((ext_vector_type(16)));
typedef _Float16 v8h  __attribute__((ext_vector_type(8)));
typedef float    v8f  __attribute__((ext_vector_type(8)));

#define BATCH 2
#define SEQ   2048
#define DIM   1024
#define NH    16
#define HD    64
#define SCALE 0.125f   // 1/sqrt(64); applied twice total, matching the reference

// ---------------------------------------------------------------------------
// WMMA wrapper: D = A(16x32 f16) x B(32x16 f16) + C(16x16 f32)
// ---------------------------------------------------------------------------
__device__ __forceinline__ v8f wmma16x16x32(v16h a, v16h b, v8f c) {
  return __builtin_amdgcn_wmma_f32_16x16x32_f16(
      /*neg_a=*/false, a, /*neg_b=*/false, b,
      /*c_mod=*/(short)0, c, /*reuse_a=*/false, /*reuse_b=*/false);
}

// Compiler-only fence: LDS ops within one wave complete in order (DScnt
// in-order); this just stops the compiler reordering the wave-private
// store->load sequence through LDS.
__device__ __forceinline__ void wave_lds_fence() { asm volatile("" ::: "memory"); }

// ---------------------------------------------------------------------------
// A-fragment (16x32, f16), row-major source [*, lda]:
//   lane l: m = l&15, khi = l>>4
//   element j<8  : K = k0 + 8*khi + j          (contiguous 16B)
//   element j>=8 : K = k0 + 16 + 8*khi + (j-8) (contiguous 16B)
// ---------------------------------------------------------------------------
__device__ __forceinline__ v16h load_a_frag(const _Float16* __restrict__ base,
                                            int lda, int m0, int k0) {
  const int l = threadIdx.x & 31;
  const int m = l & 15, khi = l >> 4;
  const _Float16* p = base + (size_t)(m0 + m) * lda + k0 + 8 * khi;
  v8h lo = *reinterpret_cast<const v8h*>(p);
  v8h hi = *reinterpret_cast<const v8h*>(p + 16);
  v16h r;
#pragma unroll
  for (int j = 0; j < 8; ++j) { r[j] = lo[j]; r[j + 8] = hi[j]; }
  return r;
}

// ---------------------------------------------------------------------------
// B-fragment (32x16, f16) for computing A x SRC^T, SRC row-major [*, ldb]:
//   B[k,n] = SRC[n0+n, k0+k].  lane l: n = l&15, khi = l>>4,
//   element j<8: K = k0 + 8*khi + j; j>=8: K = k0 + 16 + 8*khi + (j-8) (two 16B loads)
// ---------------------------------------------------------------------------
__device__ __forceinline__ v16h load_b_frag(const _Float16* __restrict__ base,
                                            int ldb, int n0, int k0) {
  const int l = threadIdx.x & 31;
  const int n = l & 15, khi = l >> 4;
  const _Float16* p = base + (size_t)(n0 + n) * ldb + k0 + 8 * khi;
  v8h lo = *reinterpret_cast<const v8h*>(p);
  v8h hi = *reinterpret_cast<const v8h*>(p + 16);
  v16h r;
#pragma unroll
  for (int j = 0; j < 8; ++j) { r[j] = lo[j]; r[j + 8] = hi[j]; }
  return r;
}

// ---------------------------------------------------------------------------
// fp32 -> f16 converter
// ---------------------------------------------------------------------------
__global__ void to_f16_kernel(const float* __restrict__ in,
                              _Float16* __restrict__ out, int n) {
  int i = blockIdx.x * blockDim.x + threadIdx.x;
  if (i < n) out[i] = (_Float16)in[i];
}

// ---------------------------------------------------------------------------
// QKV projection: y = x @ W^T  (x:[B*S,D] f16, W:[D,D] f16 row-major)
// 256-thread blocks, 8 waves; each wave computes a 32x64 output tile
// (8 accumulators = 64 VGPRs: spill-free; 12 b128 loads per 8 WMMAs/K-step).
//   which==0 -> Q (scaled by SCALE), stored [B,H,S,HD]
//   which==1 -> K, stored [B,H,S,HD]
//   which==2 -> V, stored TRANSPOSED [B,H,HD,S]
// ---------------------------------------------------------------------------
__global__ void proj_qkv_kernel(const _Float16* __restrict__ xh,
                                const _Float16* __restrict__ Wqh,
                                const _Float16* __restrict__ Wkh,
                                const _Float16* __restrict__ Wvh,
                                _Float16* __restrict__ Qh,
                                _Float16* __restrict__ Kh,
                                _Float16* __restrict__ Vt) {
  const int wave  = threadIdx.x >> 5;
  const int tile  = blockIdx.x * 8 + wave;   // 2048 tiles: 128 m-tiles x 16 n-tiles
  const int mbase = (tile >> 4) * 32;
  const int n0    = (tile & 15) * 64;
  const int which = blockIdx.z;
  const _Float16* W = (which == 0) ? Wqh : (which == 1) ? Wkh : Wvh;

  v8f acc[2][4];
#pragma unroll
  for (int m = 0; m < 2; ++m)
#pragma unroll
    for (int n = 0; n < 4; ++n) acc[m][n] = 0.f;

  for (int kk = 0; kk < DIM; kk += 32) {
    v16h a0 = load_a_frag(xh, DIM, mbase, kk);
    v16h a1 = load_a_frag(xh, DIM, mbase + 16, kk);
#pragma unroll
    for (int n = 0; n < 4; ++n) {
      const v16h bf = load_b_frag(W, DIM, n0 + 16 * n, kk);
      acc[0][n] = wmma16x16x32(a0, bf, acc[0][n]);
      acc[1][n] = wmma16x16x32(a1, bf, acc[1][n]);
    }
  }

  const int l = threadIdx.x & 31;
  const int c = l & 15, khi = l >> 4;
#pragma unroll
  for (int m = 0; m < 2; ++m) {
#pragma unroll
    for (int n = 0; n < 4; ++n) {
#pragma unroll
      for (int r = 0; r < 8; ++r) {
        const int row = mbase + 16 * m + r + 8 * khi;  // token index in [0, B*S)
        const int col = n0 + 16 * n + c;               // feature in [0, D)
        const int b_  = row >> 11;                     // / SEQ
        const int s_  = row & (SEQ - 1);
        const int h_  = col >> 6;                      // / HD
        const int hc  = col & (HD - 1);
        const float v = acc[m][n][r];
        if (which == 0)
          Qh[((size_t)((b_ * NH + h_) * SEQ + s_)) * HD + hc] = (_Float16)(v * SCALE);
        else if (which == 1)
          Kh[((size_t)((b_ * NH + h_) * SEQ + s_)) * HD + hc] = (_Float16)v;
        else
          Vt[((size_t)((b_ * NH + h_) * HD + hc)) * SEQ + s_] = (_Float16)v;
      }
    }
  }
}

// ---------------------------------------------------------------------------
// Flash attention: 4 waves per block, one wave per (b, h, 16-row query tile).
// Online softmax over 32-wide key steps; P staged through a wave-private LDS
// tile to reshape the C/D score layout into the WMMA A-fragment layout.
// No cross-wave synchronization (per-wave causal trip counts differ).
// ---------------------------------------------------------------------------
__global__ void flash_attn_kernel(const _Float16* __restrict__ Qh,
                                  const _Float16* __restrict__ Kh,
                                  const _Float16* __restrict__ Vt,
                                  _Float16* __restrict__ Ah) {
  __shared__ _Float16 pbuf_all[4][16 * 40];  // per-wave 16x32 tile, row stride 40
  const int wave = threadIdx.x >> 5;
  _Float16* pbuf = pbuf_all[wave];

  const int mt = blockIdx.x * 4 + wave;
  const int h = blockIdx.y, b = blockIdx.z;
  const int mbase = mt * 16;
  const int l = threadIdx.x & 31;
  const int c = l & 15, khi = l >> 4;

  const _Float16* Qb = Qh + (size_t)(b * NH + h) * SEQ * HD;
  const _Float16* Kb = Kh + (size_t)(b * NH + h) * SEQ * HD;
  const _Float16* Vb = Vt + (size_t)(b * NH + h) * HD * SEQ;

  // Q fragments (16 rows x HD=64) held in registers for the whole pass
  const v16h q0 = load_a_frag(Qb, HD, mbase, 0);
  const v16h q1 = load_a_frag(Qb, HD, mbase, 32);

  v8f acc[4];
#pragma unroll
  for (int t = 0; t < 4; ++t) acc[t] = 0.f;
  float rmax[8], rsum[8];
#pragma unroll
  for (int r = 0; r < 8; ++r) { rmax[r] = -1e30f; rsum[r] = 0.f; }

  const int tend = mbase + 16;  // causal: keys t <= row < mbase+16
  for (int t0 = 0; t0 < tend; t0 += 32) {
    // ---- scores: S[16x32] = q[16x64] . K[t0:t0+32, 0:64]^T, two 16-col halves
    v8f s_lo = 0.f, s_hi = 0.f;
    s_lo = wmma16x16x32(q0, load_b_frag(Kb, HD, t0, 0), s_lo);
    s_lo = wmma16x16x32(q1, load_b_frag(Kb, HD, t0, 32), s_lo);
    s_hi = wmma16x16x32(q0, load_b_frag(Kb, HD, t0 + 16, 0), s_hi);
    s_hi = wmma16x16x32(q1, load_b_frag(Kb, HD, t0 + 16, 32), s_hi);

    // ---- scale (2nd SCALE), causal mask, online softmax
    float alpha[8];
#pragma unroll
    for (int r = 0; r < 8; ++r) {
      const int row = mbase + r + 8 * khi;      // global query index
      float a0 = s_lo[r] * SCALE;
      float a1 = s_hi[r] * SCALE;
      if (t0 + c > row)      a0 = -1e30f;
      if (t0 + 16 + c > row) a1 = -1e30f;
      float mx = fmaxf(a0, a1);
#pragma unroll
      for (int off = 8; off; off >>= 1) mx = fmaxf(mx, __shfl_xor(mx, off, 16));
      const float mnew = fmaxf(rmax[r], mx);
      const float al = __expf(rmax[r] - mnew);
      rmax[r] = mnew;
      const float p0 = __expf(a0 - mnew);
      const float p1 = __expf(a1 - mnew);
      float ps = p0 + p1;
#pragma unroll
      for (int off = 8; off; off >>= 1) ps += __shfl_xor(ps, off, 16);
      rsum[r] = rsum[r] * al + ps;
      alpha[r] = al;
      const int rr = r + 8 * khi;
      pbuf[rr * 40 + c]      = (_Float16)p0;
      pbuf[rr * 40 + 16 + c] = (_Float16)p1;
    }
    wave_lds_fence();

    // rescale running accumulators
#pragma unroll
    for (int t = 0; t < 4; ++t)
#pragma unroll
      for (int r = 0; r < 8; ++r) acc[t][r] *= alpha[r];

    // reload P as an A-fragment (16x32) from LDS (wave-private; DS in-order)
    v16h pf;
    {
      const _Float16* p = &pbuf[c * 40 + 8 * khi];
      v8h lo = *reinterpret_cast<const v8h*>(p);
      v8h hi = *reinterpret_cast<const v8h*>(p + 16);
#pragma unroll
      for (int j = 0; j < 8; ++j) { pf[j] = lo[j]; pf[j + 8] = hi[j]; }
    }

    // ---- O += P[16x32] . V[t0:t0+32, 0:64]   (V stored transposed [HD,SEQ])
#pragma unroll
    for (int t = 0; t < 4; ++t)
      acc[t] = wmma16x16x32(pf, load_b_frag(Vb, SEQ, 16 * t, t0), acc[t]);
    wave_lds_fence();
  }

  // ---- normalize and store per-head output, concat-head layout [B*S, D]
  _Float16* Aout = Ah + ((size_t)(b * SEQ + mbase)) * DIM + h * HD;
#pragma unroll
  for (int r = 0; r < 8; ++r) {
    const float inv = 1.0f / rsum[r];
    const int row = r + 8 * khi;
#pragma unroll
    for (int t = 0; t < 4; ++t)
      Aout[(size_t)row * DIM + 16 * t + c] = (_Float16)(acc[t][r] * inv);
  }
}

// ---------------------------------------------------------------------------
// Output projection: out = Ah @ Wo^T + bo  (fp32 output), 32x64 per wave
// ---------------------------------------------------------------------------
__global__ void out_proj_kernel(const _Float16* __restrict__ Ah,
                                const _Float16* __restrict__ Woh,
                                const float* __restrict__ bo,
                                float* __restrict__ out) {
  const int wave  = threadIdx.x >> 5;
  const int tile  = blockIdx.x * 8 + wave;   // 2048 tiles
  const int mbase = (tile >> 4) * 32;
  const int n0    = (tile & 15) * 64;

  v8f acc[2][4];
#pragma unroll
  for (int m = 0; m < 2; ++m)
#pragma unroll
    for (int n = 0; n < 4; ++n) acc[m][n] = 0.f;

  for (int kk = 0; kk < DIM; kk += 32) {
    v16h a0 = load_a_frag(Ah, DIM, mbase, kk);
    v16h a1 = load_a_frag(Ah, DIM, mbase + 16, kk);
#pragma unroll
    for (int n = 0; n < 4; ++n) {
      const v16h bf = load_b_frag(Woh, DIM, n0 + 16 * n, kk);
      acc[0][n] = wmma16x16x32(a0, bf, acc[0][n]);
      acc[1][n] = wmma16x16x32(a1, bf, acc[1][n]);
    }
  }

  const int l = threadIdx.x & 31;
  const int c = l & 15, khi = l >> 4;
#pragma unroll
  for (int m = 0; m < 2; ++m) {
#pragma unroll
    for (int n = 0; n < 4; ++n) {
#pragma unroll
      for (int r = 0; r < 8; ++r) {
        const int row = mbase + 16 * m + r + 8 * khi;
        const int col = n0 + 16 * n + c;
        out[(size_t)row * DIM + col] = acc[m][n][r] + bo[col];
      }
    }
  }
}

// ---------------------------------------------------------------------------
// Launch
// ---------------------------------------------------------------------------
extern "C" void kernel_launch(void* const* d_in, const int* in_sizes, int n_in,
                              void* d_out, int out_size, void* d_ws, size_t ws_size,
                              hipStream_t stream) {
  (void)in_sizes; (void)n_in; (void)out_size; (void)ws_size;

  const float* x  = (const float*)d_in[0];
  const float* Wq = (const float*)d_in[1];
  const float* Wk = (const float*)d_in[2];
  const float* Wv = (const float*)d_in[3];
  const float* Wo = (const float*)d_in[4];
  const float* bo = (const float*)d_in[5];
  float* out = (float*)d_out;

  const int XN = BATCH * SEQ * DIM;  // 4,194,304
  const int WN = DIM * DIM;          // 1,048,576

  _Float16* ws  = (_Float16*)d_ws;
  _Float16* xh  = ws;
  _Float16* Wqh = xh  + XN;
  _Float16* Wkh = Wqh + WN;
  _Float16* Wvh = Wkh + WN;
  _Float16* Woh = Wvh + WN;
  _Float16* Qh  = Woh + WN;   // [B,H,S,HD], pre-scaled by SCALE
  _Float16* Kh  = Qh  + XN;   // [B,H,S,HD]
  _Float16* Vt  = Kh  + XN;   // [B,H,HD,S] (transposed)
  _Float16* Ah  = Vt  + XN;   // [B*S, D] concat-head attention output

  to_f16_kernel<<<(XN + 255) / 256, 256, 0, stream>>>(x,  xh,  XN);
  to_f16_kernel<<<(WN + 255) / 256, 256, 0, stream>>>(Wq, Wqh, WN);
  to_f16_kernel<<<(WN + 255) / 256, 256, 0, stream>>>(Wk, Wkh, WN);
  to_f16_kernel<<<(WN + 255) / 256, 256, 0, stream>>>(Wv, Wvh, WN);
  to_f16_kernel<<<(WN + 255) / 256, 256, 0, stream>>>(Wo, Woh, WN);

  // 128 m-tiles x 16 n-tiles = 2048 32x64 tiles, 8 waves per block -> 256 blocks
  proj_qkv_kernel<<<dim3(256, 1, 3), 256, 0, stream>>>(
      xh, Wqh, Wkh, Wvh, Qh, Kh, Vt);

  // SEQ/16 = 128 query tiles per (b,h); 4 waves per block -> 32 blocks in x
  flash_attn_kernel<<<dim3(SEQ / 64, NH, BATCH), 128, 0, stream>>>(Qh, Kh, Vt, Ah);

  out_proj_kernel<<<dim3(256, 1, 1), 256, 0, stream>>>(Ah, Woh, bo, out);
}
